// AffineVoxelMorph_42614665511598
// MI455X (gfx1250) — matrix operand order repaired and checked by reading, not compiled, not run on hardware
//
#include <hip/hip_runtime.h>
#include <hip/hip_bf16.h>
#include <math.h>

typedef __attribute__((ext_vector_type(16))) _Float16 v16h;
typedef __attribute__((ext_vector_type(8)))  _Float16 v8h;
typedef __attribute__((ext_vector_type(8)))  float    v8f;

#define V96 884736
#define V48 110592
#define V24 13824
#define V12 1728
#define V6  216

__device__ __forceinline__ int clampi(int v, int lo, int hi) {
  return v < lo ? lo : (v > hi ? hi : v);
}

// ---------------------------------------------------------------------------
// Weight repack, K ordered as (tap, cin): chunk = tap*KCt + sub.
// idx = ((nt*27*KCt + tap*KCt + sub)*32 + lane)*16 + i
//   -> B[k][n], k = sub*32 + (lane>>4)*16 + i (cin within padded row),
//      n = nt*16 + (lane&15).  Zero outside real Cin/Cout.
// ---------------------------------------------------------------------------
__global__ void pack_weights(const float* __restrict__ w, _Float16* __restrict__ pack,
                             int CoutR, int CinR, int KCt, int Nt) {
  int idx = blockIdx.x * blockDim.x + threadIdx.x;
  int chunks = 27 * KCt;
  int total = Nt * chunks * 512;
  if (idx >= total) return;
  int i    = idx & 15;
  int lane = (idx >> 4) & 31;
  int rest = idx >> 9;               // nt*chunks + chunk
  int chunk = rest % chunks;
  int nt    = rest / chunks;
  int tap = chunk / KCt;
  int sub = chunk - tap * KCt;
  int cin = sub * 32 + ((lane >> 4) << 4) + i;
  int n   = nt * 16 + (lane & 15);
  float v = 0.f;
  if (cin < CinR && n < CoutR) v = w[(n * CinR + cin) * 27 + tap];
  pack[idx] = (_Float16)v;
}

// ---------------------------------------------------------------------------
// Implicit-GEMM conv / conv-transpose on v_wmma_f32_16x16x32_f16.
// Activations: f16 channel-last [voxel][Pin], Pin = KCT*32.
// One wave = 16 voxels x (NT*16) out-channels; A fragments reused across NT.
// Steady state per A pair: NT WMMAs, 2 b128 A loads, NT 32B B loads.
// ---------------------------------------------------------------------------
struct ConvP {
  const _Float16* inH;
  const _Float16* pack;
  const float* bias;
  _Float16* outH;     // f16 channel-last output (pitch Pout), or
  float* outF;        // f32 channel-last pitch-3 output (fin layer)
  int inS, outS, Pout, CoutR, coff, numM, stride, pad, mode, lrelu, f32out;
};

template <int NT, int KCT>
__global__ void conv_wmma(ConvP p) {
  int lane = threadIdx.x & 31;
  int wv = blockIdx.x * (blockDim.x >> 5) + (threadIdx.x >> 5);
  if (wv >= p.numM) return;                      // whole-wave exit: EXEC stays full
  int mt = wv;
  int Vout = p.outS * p.outS * p.outS;
  int khalf = lane >> 4;
  int m = mt * 16 + (lane & 15);
  bool mok = m < Vout;
  int mc = mok ? m : (Vout - 1);
  int ox = mc % p.outS;
  int tq = mc / p.outS;
  int oy = tq % p.outS;
  int oz = tq / p.outS;
  const int inS = p.inS;
  const int P = KCT * 32;
  const v16h* packV = (const v16h*)p.pack;
  v8f acc[NT];
#pragma unroll
  for (int u = 0; u < NT; ++u) acc[u] = (v8f){};
  for (int tap = 0; tap < 27; ++tap) {
    int tz = tap / 9;
    int r = tap - 9 * tz;
    int ty = r / 3;
    int tx = r - 3 * ty;
    int iz, iy, ix;
    bool ok;
    if (p.mode == 0) {
      iz = oz * p.stride + tz - p.pad;
      iy = oy * p.stride + ty - p.pad;
      ix = ox * p.stride + tx - p.pad;
      ok = (unsigned)iz < (unsigned)inS && (unsigned)iy < (unsigned)inS &&
           (unsigned)ix < (unsigned)inS;
    } else {  // conv-transpose s=2, gather form: x[(o+1-t)/2], parity-gated
      int nz = oz + 1 - tz, ny = oy + 1 - ty, nx = ox + 1 - tx;
      ok = ((nz | ny | nx) >= 0) && !((nz | ny | nx) & 1);
      iz = nz >> 1; iy = ny >> 1; ix = nx >> 1;
      ok = ok && iz < inS && iy < inS && ix < inS;
    }
    ok = ok && mok;
    const _Float16* row = p.inH + (long)((iz * inS + iy) * inS + ix) * P;
    if (tap + 1 < 27)   // keep next tap's B fragments hot (global_prefetch_b8)
      __builtin_prefetch((const void*)(packV + (size_t)(tap + 1) * KCT * 32 + lane), 0, 1);
#pragma unroll
    for (int sub = 0; sub < KCT; ++sub) {
      v16h a = {};
      if (ok) {
        // A in contiguous k order (matches B pack): lane holds K [khalf*16,+16)
        const v8h* rp = (const v8h*)(row + sub * 32 + khalf * 16);
        v8h lo = rp[0];
        v8h hi = rp[1];
#pragma unroll
        for (int i = 0; i < 8; ++i) { a[i] = lo[i]; a[8 + i] = hi[i]; }
      }
#pragma unroll
      for (int u = 0; u < NT; ++u) {
        v16h b = packV[((u * 27 + tap) * KCT + sub) * 32 + lane];
        acc[u] = __builtin_amdgcn_wmma_f32_16x16x32_f16(false, a, false, b, (short)0,
                                                        acc[u], false, false);
      }
    }
  }
  // D layout: element i -> M = i + 8*khalf, N = lane&15 (+16 per N tile)
  if (p.f32out) {
#pragma unroll
    for (int u = 0; u < NT; ++u) {
      int ng = u * 16 + (lane & 15);
      if (ng < p.CoutR) {
        float bv = p.bias[ng];
#pragma unroll
        for (int i = 0; i < 8; ++i) {
          int mm = mt * 16 + i + 8 * khalf;
          if (mm < Vout) p.outF[(size_t)mm * 3 + ng] = acc[u][i] + bv;
        }
      }
    }
  } else {
#pragma unroll
    for (int u = 0; u < NT; ++u) {
      int ng = u * 16 + (lane & 15);
      float bv = (ng < p.CoutR) ? p.bias[ng] : 0.f;
#pragma unroll
      for (int i = 0; i < 8; ++i) {
        int mm = mt * 16 + i + 8 * khalf;
        if (mm < Vout) {
          float v = (ng < p.CoutR) ? acc[u][i] + bv : 0.f;
          if (p.lrelu && v < 0.f) v *= 0.2f;
          p.outH[(size_t)mm * p.Pout + p.coff + ng] = (_Float16)v;
        }
      }
    }
  }
}

// ---------------------------------------------------------------------------
// Layout / glue kernels
// ---------------------------------------------------------------------------
__global__ void make_x16(const float* __restrict__ s, const float* __restrict__ t,
                         _Float16* __restrict__ x) {
  int v = blockIdx.x * blockDim.x + threadIdx.x;
  if (v >= V96) return;
  _Float16* rw = x + (size_t)v * 32;
  rw[0] = (_Float16)s[v];
  rw[1] = (_Float16)t[v];
#pragma unroll
  for (int c = 2; c < 32; ++c) rw[c] = (_Float16)0.f;
}

__global__ void concat_copy(const _Float16* __restrict__ src, _Float16* __restrict__ dst,
                            int V, int srcP, int dstP, int dstOff, int copyC, int totC) {
  int idx = blockIdx.x * blockDim.x + threadIdx.x;
  int c = idx % totC;
  int v = idx / totC;
  if (v >= V) return;
  dst[(size_t)v * dstP + dstOff + c] =
      (c < copyC) ? src[(size_t)v * srcP + c] : (_Float16)0.f;
}

__global__ void zfill(_Float16* __restrict__ dst, int V, int P, int c0) {
  int nc = P - c0;
  int idx = blockIdx.x * blockDim.x + threadIdx.x;
  int c = idx % nc;
  int v = idx / nc;
  if (v >= V) return;
  dst[(size_t)v * P + c0 + c] = (_Float16)0.f;
}

__global__ void mean7_kernel(const _Float16* __restrict__ x, float* __restrict__ out) {
  int c = threadIdx.x;
  if (c >= 7) return;
  float s = 0.f;
  for (int v = 0; v < V6; ++v) s += (float)x[(size_t)v * 32 + c];
  out[c] = s * (1.f / 216.f);
}

__device__ __forceinline__ void mm4(const float* A, const float* B, float* C) {
  for (int i = 0; i < 4; ++i)
    for (int j = 0; j < 4; ++j) {
      float s = 0.f;
      for (int k = 0; k < 4; ++k) s += A[i * 4 + k] * B[k * 4 + j];
      C[i * 4 + j] = s;
    }
}

// expm (CSO basis) via scaling 2^-8 + 12-term Taylor + 8 squarings, then
// center-shift conjugation A' = T(+48) * expm(M) * T(-48).
__global__ void expm_affine_kernel(const float* __restrict__ p, float* __restrict__ lo) {
  if (threadIdx.x != 0 || blockIdx.x != 0) return;
  float M[16];
  for (int i = 0; i < 16; ++i) M[i] = 0.f;
  M[3] = p[0]; M[7] = p[1]; M[11] = p[2];
  M[1] = p[3]; M[4] = -p[3];
  M[2] = p[4]; M[8] = -p[4];
  M[6] = p[5]; M[9] = -p[5];
  M[0] += p[6]; M[5] += p[6]; M[10] += p[6];
  for (int i = 0; i < 16; ++i) M[i] *= (1.f / 256.f);
  float E[16], T[16], Tn[16];
  for (int i = 0; i < 16; ++i) { E[i] = (i % 5 == 0) ? 1.f : 0.f; T[i] = E[i]; }
  for (int k = 1; k <= 12; ++k) {
    mm4(T, M, Tn);
    float inv = 1.f / (float)k;
    for (int i = 0; i < 16; ++i) { T[i] = Tn[i] * inv; E[i] += T[i]; }
  }
  for (int s = 0; s < 8; ++s) {
    mm4(E, E, Tn);
    for (int i = 0; i < 16; ++i) E[i] = Tn[i];
  }
  float Tm[16], Tp[16], A1[16], A2[16];
  for (int i = 0; i < 16; ++i) { Tm[i] = (i % 5 == 0) ? 1.f : 0.f; Tp[i] = Tm[i]; }
  Tm[3] = -48.f; Tm[7] = -48.f; Tm[11] = -48.f;
  Tp[3] =  48.f; Tp[7] =  48.f; Tp[11] =  48.f;
  mm4(E, Tm, A1);
  mm4(Tp, A1, A2);
  for (int i = 0; i < 3; ++i)
    for (int j = 0; j < 3; ++j) lo[i * 3 + j] = A2[i * 4 + j];
  for (int i = 0; i < 3; ++i) lo[9 + i] = A2[i * 4 + 3];
}

// channel-last [S^3][3] clamped trilinear
__device__ __forceinline__ void trilerp3_cl(const float* f, int S, float g0, float g1,
                                            float g2, float* o0, float* o1, float* o2) {
  float f0 = floorf(g0), f1 = floorf(g1), f2 = floorf(g2);
  float w0 = g0 - f0, w1 = g1 - f1, w2 = g2 - f2;
  int i0 = (int)f0, i1 = (int)f1, i2 = (int)f2;
  float a0 = 0.f, a1 = 0.f, a2 = 0.f;
  for (int d0 = 0; d0 < 2; ++d0)
    for (int d1 = 0; d1 < 2; ++d1)
      for (int d2 = 0; d2 < 2; ++d2) {
        int x = clampi(i0 + d0, 0, S - 1);
        int y = clampi(i1 + d1, 0, S - 1);
        int z = clampi(i2 + d2, 0, S - 1);
        float w = (d0 ? w0 : 1.f - w0) * (d1 ? w1 : 1.f - w1) * (d2 ? w2 : 1.f - w2);
        const float* ptr = f + ((size_t)((x * S + y) * S + z)) * 3;
        a0 += w * ptr[0]; a1 += w * ptr[1]; a2 += w * ptr[2];
      }
  *o0 = a0; *o1 = a1; *o2 = a2;
}

__device__ __forceinline__ float trilerp1(const float* f, int S,
                                          float g0, float g1, float g2) {
  float f0 = floorf(g0), f1 = floorf(g1), f2 = floorf(g2);
  float w0 = g0 - f0, w1 = g1 - f1, w2 = g2 - f2;
  int i0 = (int)f0, i1 = (int)f1, i2 = (int)f2;
  float a = 0.f;
  for (int d0 = 0; d0 < 2; ++d0)
    for (int d1 = 0; d1 < 2; ++d1)
      for (int d2 = 0; d2 < 2; ++d2) {
        int x = clampi(i0 + d0, 0, S - 1);
        int y = clampi(i1 + d1, 0, S - 1);
        int z = clampi(i2 + d2, 0, S - 1);
        float w = (d0 ? w0 : 1.f - w0) * (d1 ? w1 : 1.f - w1) * (d2 ? w2 : 1.f - w2);
        a += w * f[(x * S + y) * S + z];
      }
  return a;
}

// vel (f32 channel-last 96^3) -> half-res field premultiplied by 0.5/2^8
__global__ void resize_down_kernel(const float* __restrict__ vel,
                                   float* __restrict__ d, float scale) {
  int v = blockIdx.x * blockDim.x + threadIdx.x;
  if (v >= V48) return;
  int c2 = v % 48;
  int t = v / 48;
  int c1 = t % 48;
  int c0 = t / 48;
  const float sc = 95.f / 47.f;
  float a, b, c;
  trilerp3_cl(vel, 96, c0 * sc, c1 * sc, c2 * sc, &a, &b, &c);
  d[v * 3 + 0] = a * scale;
  d[v * 3 + 1] = b * scale;
  d[v * 3 + 2] = c * scale;
}

__global__ void sq_step_kernel(const float* __restrict__ din, float* __restrict__ dout) {
  int v = blockIdx.x * blockDim.x + threadIdx.x;
  if (v >= V48) return;
  int c2 = v % 48;
  int t = v / 48;
  int c1 = t % 48;
  int c0 = t / 48;
  float d0 = din[v * 3 + 0], d1 = din[v * 3 + 1], d2 = din[v * 3 + 2];
  float a, b, c;
  trilerp3_cl(din, 48, c0 + d0, c1 + d1, c2 + d2, &a, &b, &c);
  dout[v * 3 + 0] = d0 + a;
  dout[v * 3 + 1] = d1 + b;
  dout[v * 3 + 2] = d2 + c;
}

__global__ void make_grid_kernel(const float* __restrict__ d, float* __restrict__ grid) {
  int v = blockIdx.x * blockDim.x + threadIdx.x;
  if (v >= V96) return;
  int c2 = v % 96;
  int t = v / 96;
  int c1 = t % 96;
  int c0 = t / 96;
  const float sc = 47.f / 95.f;
  float g0 = c0 * sc, g1 = c1 * sc, g2 = c2 * sc;
  float a, b, c;
  trilerp3_cl(d, 48, g0, g1, g2, &a, &b, &c);
  grid[v * 3 + 0] = 2.f * (g0 + a);
  grid[v * 3 + 1] = 2.f * (g1 + b);
  grid[v * 3 + 2] = 2.f * (g2 + c);
}

__global__ void affine_kernel(float* __restrict__ grid, const float* __restrict__ lo) {
  int v = blockIdx.x * blockDim.x + threadIdx.x;
  if (v >= V96) return;
  float g0 = grid[v * 3 + 0], g1 = grid[v * 3 + 1], g2 = grid[v * 3 + 2];
  grid[v * 3 + 0] = lo[0] * g0 + lo[1] * g1 + lo[2] * g2 + lo[9];
  grid[v * 3 + 1] = lo[3] * g0 + lo[4] * g1 + lo[5] * g2 + lo[10];
  grid[v * 3 + 2] = lo[6] * g0 + lo[7] * g1 + lo[8] * g2 + lo[11];
}

__global__ void warp_kernel(const float* __restrict__ src, const float* __restrict__ grid,
                            float* __restrict__ out) {
  int v = blockIdx.x * blockDim.x + threadIdx.x;
  if (v >= V96) return;
  float g0 = grid[v * 3 + 0], g1 = grid[v * 3 + 1], g2 = grid[v * 3 + 2];
  float s = trilerp1(src, 96, g0, g1, g2);
  bool inb = g0 >= 0.f && g0 <= 95.f && g1 >= 0.f && g1 <= 95.f &&
             g2 >= 0.f && g2 <= 95.f;
  out[v] = inb ? s : 0.f;
}

// ---------------------------------------------------------------------------
// Host orchestration
// ---------------------------------------------------------------------------
extern "C" void kernel_launch(void* const* d_in, const int* in_sizes, int n_in,
                              void* d_out_v, int out_size, void* d_ws, size_t ws_size,
                              hipStream_t stream) {
  (void)in_sizes; (void)n_in; (void)out_size; (void)ws_size;
  const float* src = (const float*)d_in[0];
  const float* tgt = (const float*)d_in[1];
  float* out = (float*)d_out_v;
  float* vel = out + V96;                     // [V96][3] f32 channel-last

  // layer t: w = d_in[2+2t], b = d_in[3+2t]
  const int CO[18]  = {16,32,32,32, 7, 16,32,32,32, 32,32,32,16, 32,32,32,16, 3};
  const int CI[18]  = { 2,16,32,32,32,  2,16,32,32, 32,32,32,32, 64,64,48,18, 16};
  const int PIN[18] = {32,32,32,32,32, 32,32,32,32, 32,32,32,32, 64,64,64,32, 32};

  char* ws = (char*)d_ws;
  size_t off = 0;
  auto alloc = [&](size_t bytes) -> void* {
    void* p = ws + off;
    off += (bytes + 63) & ~(size_t)63;
    return p;
  };

  _Float16* X16  = (_Float16*)alloc((size_t)V96 * 32 * 2);   // concat(src,tgt), ch 2..31 = 0
  _Float16* E1   = (_Float16*)alloc((size_t)V48 * 32 * 2);
  _Float16* E2   = (_Float16*)alloc((size_t)V24 * 32 * 2);
  _Float16* E3   = (_Float16*)alloc((size_t)V12 * 32 * 2);
  _Float16* E4   = (_Float16*)alloc((size_t)V6  * 32 * 2);
  _Float16* R1   = (_Float16*)alloc((size_t)V48 * 32 * 2);   // CNN-head rotating
  _Float16* R2   = (_Float16*)alloc((size_t)V24 * 32 * 2);
  _Float16* BigA = (_Float16*)alloc((size_t)V96 * 32 * 2);   // concat inputs
  _Float16* BigB = (_Float16*)alloc((size_t)V96 * 32 * 2);   // dec outputs / f32 grid
  float*    D0   = (float*)alloc((size_t)3 * V48 * 4);
  float*    D1   = (float*)alloc((size_t)3 * V48 * 4);
  float*    LO   = (float*)alloc(64);

  _Float16* pk[18];
  int KCT[18], NT[18];
  for (int t = 0; t < 18; ++t) {
    KCT[t] = PIN[t] / 32;
    NT[t] = (CO[t] + 15) / 16;
    pk[t] = (_Float16*)alloc((size_t)NT[t] * 27 * KCT[t] * 512 * 2);
  }

  for (int t = 0; t < 18; ++t) {
    const float* w = (const float*)d_in[2 + 2 * t];
    int total = NT[t] * 27 * KCT[t] * 512;
    pack_weights<<<(total + 255) / 256, 256, 0, stream>>>(w, pk[t], CO[t], CI[t],
                                                          KCT[t], NT[t]);
  }

  make_x16<<<(V96 + 255) / 256, 256, 0, stream>>>(src, tgt, X16);

  auto conv = [&](const _Float16* in, int t, _Float16* oH, float* oF, int Pout,
                  int coff, int inS, int outS, int stride, int pad, int mode,
                  int lrelu) {
    ConvP p;
    p.inH = in; p.pack = pk[t]; p.bias = (const float*)d_in[3 + 2 * t];
    p.outH = oH; p.outF = oF;
    p.inS = inS; p.outS = outS; p.Pout = Pout;
    p.CoutR = CO[t]; p.coff = coff;
    int Vout = outS * outS * outS;
    p.numM = (Vout + 15) / 16;
    p.stride = stride; p.pad = pad; p.mode = mode; p.lrelu = lrelu;
    p.f32out = (oF != nullptr);
    int blocks = (p.numM + 3) / 4;
    if (NT[t] == 1 && KCT[t] == 1)
      conv_wmma<1, 1><<<blocks, 128, 0, stream>>>(p);
    else if (NT[t] == 2 && KCT[t] == 1)
      conv_wmma<2, 1><<<blocks, 128, 0, stream>>>(p);
    else
      conv_wmma<2, 2><<<blocks, 128, 0, stream>>>(p);
  };
  auto fill = [&](_Float16* dst, int V, int P, int c0) {
    int total = V * (P - c0);
    zfill<<<(total + 255) / 256, 256, 0, stream>>>(dst, V, P, c0);
  };
  auto ccopy = [&](const _Float16* s, _Float16* d, int V, int sP, int dP, int dOff,
                   int copyC, int totC) {
    int total = V * totC;
    concat_copy<<<(total + 255) / 256, 256, 0, stream>>>(s, d, V, sP, dP, dOff,
                                                         copyC, totC);
  };

  // ---- CNN affine head (stride-2 SAME: pad_lo = 0) ----
  conv(X16, 0, R1, nullptr, 32, 0, 96, 48, 2, 0, 0, 1);  fill(R1, V48, 32, 16);
  conv(R1,  1, R2, nullptr, 32, 0, 48, 24, 2, 0, 0, 1);
  conv(R2,  2, R1, nullptr, 32, 0, 24, 12, 2, 0, 0, 1);
  conv(R1,  3, R2, nullptr, 32, 0, 12,  6, 2, 0, 0, 1);
  conv(R2,  4, R1, nullptr, 32, 0,  6,  6, 1, 1, 0, 0);
  mean7_kernel<<<1, 32, 0, stream>>>(R1, out + (size_t)4 * V96);
  expm_affine_kernel<<<1, 32, 0, stream>>>(out + (size_t)4 * V96, LO);

  // ---- encoder ----
  conv(X16, 5, E1, nullptr, 32, 0, 96, 48, 2, 0, 0, 1);  fill(E1, V48, 32, 16);
  conv(E1,  6, E2, nullptr, 32, 0, 48, 24, 2, 0, 0, 1);
  conv(E2,  7, E3, nullptr, 32, 0, 24, 12, 2, 0, 0, 1);
  conv(E3,  8, E4, nullptr, 32, 0, 12,  6, 2, 0, 0, 1);

  // ---- decoder: upconv -> concat skip -> conv ----
  conv(E4,   9, BigA, nullptr, 64, 0,  6, 12, 2, 1, 1, 1);     // up0 -> CAT0[0:32)
  ccopy(E3, BigA, V12, 32, 64, 32, 32, 32);                    // CAT0[32:64) = E3
  conv(BigA, 13, BigB, nullptr, 32, 0, 12, 12, 1, 1, 0, 1);    // dec0
  conv(BigB, 10, BigA, nullptr, 64, 0, 12, 24, 2, 1, 1, 1);    // up1
  ccopy(E2, BigA, V24, 32, 64, 32, 32, 32);                    // CAT1[32:64) = E2
  conv(BigA, 14, BigB, nullptr, 32, 0, 24, 24, 1, 1, 0, 1);    // dec1
  conv(BigB, 11, BigA, nullptr, 64, 0, 24, 48, 2, 1, 1, 1);    // up2
  ccopy(E1, BigA, V48, 32, 64, 32, 16, 32);                    // CAT2[32:48)=E1, [48:64)=0
  conv(BigA, 15, BigB, nullptr, 32, 0, 48, 48, 1, 1, 0, 1);    // dec2
  conv(BigB, 12, BigA, nullptr, 32, 0, 48, 96, 2, 1, 1, 1);    // up3 -> CAT3[0:16)
  ccopy(X16, BigA, V96, 32, 32, 16, 2, 16);                    // CAT3[16:18)=x, [18:32)=0
  conv(BigA, 16, BigB, nullptr, 32, 0, 96, 96, 1, 1, 0, 1);    // dec3 -> [0:16)
  fill(BigB, V96, 32, 16);                                     // pad ch 16..31 for fin
  conv(BigB, 17, nullptr, vel, 0, 0, 96, 96, 1, 1, 0, 0);      // fin -> vel (f32, pitch 3)

  // ---- velocity integration, grid, affine, warp ----
  resize_down_kernel<<<(V48 + 255) / 256, 256, 0, stream>>>(vel, D0, 0.5f / 256.f);
  for (int i = 0; i < 8; ++i) {
    const float* di = (i & 1) ? D1 : D0;
    float* dn = (i & 1) ? D0 : D1;
    sq_step_kernel<<<(V48 + 255) / 256, 256, 0, stream>>>(di, dn);
  }
  float* gridF = (float*)BigB;                                 // DEC3 dead after fin
  make_grid_kernel<<<(V96 + 255) / 256, 256, 0, stream>>>(D0, gridF);
  affine_kernel<<<(V96 + 255) / 256, 256, 0, stream>>>(gridF, LO);
  warp_kernel<<<(V96 + 255) / 256, 256, 0, stream>>>(src, gridF, out);
}
